// ParticleNet_8134668058717
// MI455X (gfx1250) — compile-verified, matrix operand order checked by reading, not run on hardware
//
#include <hip/hip_runtime.h>

// ---------------------------------------------------------------------------
// ParticleNet inference for MI455X (gfx1250, wave32, WMMA).
// Pipeline per block: knn -> node GEMM (U|V) -> edge assemble (+BN stats)
//   -> edge GEMM x2 (BN+ReLU fused in prologue, stats in epilogue)
//   -> BN+ReLU+mean-over-K reduce into concatenated feature buffer.
// GEMMs use v_wmma_f32_16x16x32_bf16, fp32 accumulation; activations stored
// bf16 (halves HBM traffic; 23.3 TB/s roofline ~ balanced vs WMMA compute).
// Branch-free, software-pipelined staging: feature buffer stride padded to
// 480 (zeroed) so BASEOF + ceil32(K) == 480 for all blocks; weight buffers
// zero-padded to Kpad rows. Global loads for tile kt+1 are issued before the
// WMMAs of tile kt to hide HBM latency behind compute.
// ---------------------------------------------------------------------------

typedef __attribute__((ext_vector_type(16))) __bf16 v16bf;
typedef __attribute__((ext_vector_type(8)))  float  v8f;

#define BN 64
#define BK 32
#define APITCH 40   // bf16 elems per LDS row (80B pitch -> conflict-free frags)
#define BPITCH 40

#define NODES 16384          // B*N = 256*64
#define NEDGE 262144         // B*N*K
#define FTOT  455            // real channels: 256+128+64+7
#define FSTRIDE 480          // padded stride; channels [455,480) stay zero

// ------------------------------ GEMM (WMMA) --------------------------------
// MODE 0: A is f32 (padded feats, stride lda), K = Kpad (mult 32) [node GEMM]
// MODE 1: A is bf16 pre-BN, lda==K, fused BN+ReLU prologue        [edge GEMM]
template<int TBM, int MODE>
__global__ __launch_bounds__(256) void pn_wmma_gemm(
    const void* __restrict__ Ain, long lda,
    const __bf16* __restrict__ W, const float* __restrict__ bias,
    const float* __restrict__ scale, const float* __restrict__ shift,
    __bf16* __restrict__ Out, int K, int N,
    float* __restrict__ statSum, float* __restrict__ statSq, int do_stats)
{
    constexpr int TM = TBM / 64;             // 16-row tiles per wave (M dir)
    __shared__ __align__(16) __bf16 As[TBM * APITCH];
    __shared__ __align__(16) __bf16 Bs[BN * BPITCH];   // transposed [n][k]
    __shared__ float lsum[BN];
    __shared__ float lsq[BN];

    const int tid  = threadIdx.x;
    const int lane = tid & 31;
    const int wave = tid >> 5;
    const int wm   = wave >> 1;              // 0..3 (M direction)
    const int wn   = wave & 1;               // 0..1 (N direction)
    const int sel  = lane >> 4;              // 0/1 half-wave select
    const int r16  = lane & 15;
    const long m_blk = (long)blockIdx.x * TBM;
    const int  n_blk = blockIdx.y * BN;
    const int  bk    = tid >> 3;             // 0..31  (B staging row)
    const int  bn0   = (tid & 7) * 8;        // 0..56  (B staging col group)

    if (do_stats && tid < BN) { lsum[tid] = 0.f; lsq[tid] = 0.f; }

    v8f acc[TM][2] = {};
    uint4 ra[4], rbv;

    // ---- branch-free register staging (2-stage software pipeline) ----
    auto load_tile = [&](int k0) {
        if constexpr (MODE == 1) {
            const uint4* ap = (const uint4*)((const __bf16*)Ain +
                                             (m_blk + tid) * (long)K + k0);
            #pragma unroll
            for (int i = 0; i < 4; ++i) ra[i] = ap[i];          // 32 bf16
        } else {
            const uint4* ap = (const uint4*)((const float*)Ain +
                 (m_blk + (tid >> 1)) * lda + k0 + (tid & 1) * 16);
            #pragma unroll
            for (int i = 0; i < 4; ++i) ra[i] = ap[i];          // 16 f32
        }
        rbv = *(const uint4*)&W[(long)(k0 + bk) * N + n_blk + bn0]; // 8 bf16
    };

    auto store_tile = [&](int k0) {
        if constexpr (MODE == 1) {
            union { uint4 u4[4]; __bf16 b[32]; } t;
            #pragma unroll
            for (int i = 0; i < 4; ++i) t.u4[i] = ra[i];
            #pragma unroll
            for (int e = 0; e < 32; ++e) {          // scale/shift idx uniform
                float v = (float)t.b[e] * scale[k0 + e] + shift[k0 + e];
                t.b[e] = (__bf16)(v > 0.f ? v : 0.f);
            }
            uint4* d = (uint4*)&As[tid * APITCH];
            #pragma unroll
            for (int i = 0; i < 4; ++i) d[i] = t.u4[i];
        } else {
            union { uint4 u4[4]; float f[16]; } t;
            #pragma unroll
            for (int i = 0; i < 4; ++i) t.u4[i] = ra[i];
            union { uint4 u4[2]; __bf16 b[16]; } o;
            #pragma unroll
            for (int e = 0; e < 16; ++e) o.b[e] = (__bf16)t.f[e];
            uint4* d = (uint4*)&As[(tid >> 1) * APITCH + (tid & 1) * 16];
            d[0] = o.u4[0]; d[1] = o.u4[1];
        }
        union { uint4 u4; __bf16 b[8]; } wb;
        wb.u4 = rbv;
        #pragma unroll
        for (int j = 0; j < 8; ++j) Bs[(bn0 + j) * BPITCH + bk] = wb.b[j];
    };

    const int kTiles = K / BK;               // K always a multiple of 32
    load_tile(0);
    for (int kt = 0; kt < kTiles; ++kt) {
        __syncthreads();                     // LDS consumed by prev compute
        store_tile(kt * BK);
        if (kt + 1 < kTiles) load_tile((kt + 1) * BK);  // hide HBM latency
        __syncthreads();

        // ---- fragments from LDS (layout per ISA 7.12.2) + WMMA ----
        v16bf afrag[TM], bfrag[2];
        #pragma unroll
        for (int tm = 0; tm < TM; ++tm) {
            const __bf16* src = &As[(wm * (TM * 16) + tm * 16 + r16) * APITCH + sel * 8];
            uint4* dst = reinterpret_cast<uint4*>(&afrag[tm]);
            dst[0] = *reinterpret_cast<const uint4*>(src);
            dst[1] = *reinterpret_cast<const uint4*>(src + 16);
        }
        #pragma unroll
        for (int tn = 0; tn < 2; ++tn) {
            const __bf16* src = &Bs[(wn * 32 + tn * 16 + r16) * BPITCH + sel * 8];
            uint4* dst = reinterpret_cast<uint4*>(&bfrag[tn]);
            dst[0] = *reinterpret_cast<const uint4*>(src);
            dst[1] = *reinterpret_cast<const uint4*>(src + 16);
        }
        #pragma unroll
        for (int tm = 0; tm < TM; ++tm)
            #pragma unroll
            for (int tn = 0; tn < 2; ++tn)
                acc[tm][tn] = __builtin_amdgcn_wmma_f32_16x16x32_bf16(
                    false, afrag[tm], false, bfrag[tn],
                    (short)0, acc[tm][tn], false, false);
    }

    // ---- epilogue: bias, bf16 store, per-channel stats ----
    #pragma unroll
    for (int tm = 0; tm < TM; ++tm) {
        #pragma unroll
        for (int tn = 0; tn < 2; ++tn) {
            const int ncl  = wn * 32 + tn * 16 + r16;  // 0..63 local col
            const int ncol = n_blk + ncl;
            const float bv = bias ? bias[ncol] : 0.f;
            float s = 0.f, q = 0.f;
            #pragma unroll
            for (int r = 0; r < 8; ++r) {
                const long mrow = m_blk + wm * (TM * 16) + tm * 16 + sel * 8 + r;
                const float v = acc[tm][tn][r] + bv;
                Out[mrow * N + ncol] = (__bf16)v;
                s += v; q += v * v;
            }
            if (do_stats) { atomicAdd(&lsum[ncl], s); atomicAdd(&lsq[ncl], q); }
        }
    }
    if (do_stats) {
        __syncthreads();
        if (tid < BN) {
            atomicAdd(&statSum[n_blk + tid], lsum[tid]);
            atomicAdd(&statSq [n_blk + tid], lsq [tid]);
        }
    }
}

// ------------------------------ kNN ----------------------------------------
#define MAXC 200
__global__ __launch_bounds__(64) void pn_knn(
    const float* __restrict__ feat, int C, int stride, int* __restrict__ idx)
{
    __shared__ float sf[64 * MAXC];
    const int jet = blockIdx.x;
    const int tid = threadIdx.x;
    const float* fb = feat + (long)jet * 64 * stride;
    for (int t = tid; t < 64 * C; t += 64) {
        const int n = t / C, c = t % C;
        sf[n * C + c] = fb[(long)n * stride + c];
    }
    __syncthreads();

    float bd[16]; int bi[16];
    #pragma unroll
    for (int r = 0; r < 16; ++r) { bd[r] = 3.0e38f; bi[r] = 0; }

    const float* mine = &sf[tid * C];
    for (int j = 0; j < 64; ++j) {
        if (j == tid) continue;                 // loop=False: exclude self
        const float* oth = &sf[j * C];
        float d2 = 0.f;
        for (int c = 0; c < C; ++c) { const float d = mine[c] - oth[c]; d2 += d * d; }
        if (d2 < bd[15]) {
            bd[15] = d2; bi[15] = j;
            #pragma unroll
            for (int r = 15; r > 0; --r) {
                if (bd[r] < bd[r - 1]) {
                    float td = bd[r]; bd[r] = bd[r-1]; bd[r-1] = td;
                    int   ti = bi[r]; bi[r] = bi[r-1]; bi[r-1] = ti;
                }
            }
        }
    }
    int* op = idx + ((long)jet * 64 + tid) * 16;
    #pragma unroll
    for (int r = 0; r < 16; ++r) op[r] = bi[r];
}

// --------------------- edge assemble: h0 = U_i + V_j -----------------------
__global__ __launch_bounds__(256) void pn_edge_assemble(
    const __bf16* __restrict__ UV, const int* __restrict__ idx,
    __bf16* __restrict__ h, int ls,
    float* __restrict__ statSum, float* __restrict__ statSq)
{
    __shared__ float lsum[256], lsq[256];
    const int tid = threadIdx.x;
    if (tid < ls) { lsum[tid] = 0.f; lsq[tid] = 0.f; }
    __syncthreads();

    const long e0 = (long)blockIdx.x * 64;
    const int twols = 2 * ls;
    const int c = tid % ls;                 // ls divides 256 -> constant per thread
    float s = 0.f, q = 0.f;
    for (int t = tid; t < 64 * ls; t += 256) {
        const long e = e0 + (t / ls);
        const long inode = e >> 4;
        const long jet   = inode >> 6;
        const long jnode = jet * 64 + idx[e];
        const float v = (float)UV[inode * twols + c]
                      + (float)UV[jnode * twols + ls + c];
        h[e * ls + c] = (__bf16)v;
        s += v; q += v * v;
    }
    atomicAdd(&lsum[c], s); atomicAdd(&lsq[c], q);
    __syncthreads();
    if (tid < ls) {
        atomicAdd(&statSum[tid], lsum[tid]);
        atomicAdd(&statSq [tid], lsq [tid]);
    }
}

// ------------------- BN stats -> (scale, shift) -----------------------------
__global__ void pn_finalize_stats(
    const float* __restrict__ sum, const float* __restrict__ sq,
    const float* __restrict__ gamma, const float* __restrict__ beta,
    float* __restrict__ scale, float* __restrict__ shift, int n, float inv_cnt)
{
    const int c = blockIdx.x * blockDim.x + threadIdx.x;
    if (c < n) {
        const float mu  = sum[c] * inv_cnt;
        float var = sq[c] * inv_cnt - mu * mu;
        var = var < 0.f ? 0.f : var;
        const float sc = gamma[c] * rsqrtf(var + 1e-5f);
        scale[c] = sc;
        shift[c] = beta[c] - mu * sc;
    }
}

__global__ void pn_zero(float* __restrict__ p, int n)
{
    const int i = blockIdx.x * blockDim.x + threadIdx.x;
    if (i < n) p[i] = 0.f;
}

// -------- BN+ReLU + mean over K neighbors -> concat feature buffer ---------
__global__ __launch_bounds__(64) void pn_edge_reduce(
    const __bf16* __restrict__ h, const float* __restrict__ scale,
    const float* __restrict__ shift, float* __restrict__ feats, int ls, int off)
{
    const long node = blockIdx.x;
    for (int c = threadIdx.x; c < ls; c += 64) {
        const float sc = scale[c], sh = shift[c];
        float acc = 0.f;
        const __bf16* hp = h + (node * 16) * ls + c;
        #pragma unroll
        for (int kk = 0; kk < 16; ++kk) {
            const float v = (float)hp[(long)kk * ls] * sc + sh;
            acc += v > 0.f ? v : 0.f;
        }
        feats[node * FSTRIDE + off + c] = acc * (1.f / 16.f);
    }
}

// -------------------------- weight prep ------------------------------------
// Layer-0 split: Wcat[k, 0:ls] = W1-W2 ; Wcat[k, ls:2ls] = W2 ; b2 = [b | 0]
// Rows k in [C, Kpad) are zero-filled so GEMM staging is branch-free.
__global__ void pn_prep_wcat(
    const float* __restrict__ W, const float* __restrict__ b,
    __bf16* __restrict__ Wc, float* __restrict__ b2, int C, int Kpad, int ls)
{
    const int i = blockIdx.x * blockDim.x + threadIdx.x;
    const int twols = 2 * ls;
    if (i < Kpad * twols) {
        const int k = i / twols, n = i % twols;
        float val = 0.f;
        if (k < C) {
            const float w1 = W[(long)k * ls + (n % ls)];
            const float w2 = W[(long)(C + k) * ls + (n % ls)];
            val = (n < ls) ? (w1 - w2) : w2;
        }
        Wc[i] = (__bf16)val;
    }
    if (i < twols) b2[i] = (i < ls) ? b[i] : 0.f;
}

__global__ void pn_f32_to_bf16(const float* __restrict__ in, __bf16* __restrict__ out, int n)
{
    const int i = blockIdx.x * blockDim.x + threadIdx.x;
    if (i < n) out[i] = (__bf16)in[i];
}

__global__ void pn_init_feats(const float* __restrict__ x, float* __restrict__ feats)
{
    const long i = (long)blockIdx.x * blockDim.x + threadIdx.x;   // over 16384*7
    if (i < (long)NODES * 7) {
        const long node = i / 7; const int f = i % 7;
        feats[node * FSTRIDE + 448 + f] = x[i];    // raw x in channels 448..454
    }
}

// ------------------------------ head ---------------------------------------
__global__ __launch_bounds__(256) void pn_head(
    const float* __restrict__ feats, const float* __restrict__ W1,
    const float* __restrict__ b1, const float* __restrict__ W2,
    const float* __restrict__ b2, float* __restrict__ out)
{
    __shared__ float pooled[FTOT];
    __shared__ float hfc[256];
    const int b = blockIdx.x, tid = threadIdx.x;
    const float* fb = feats + (long)b * 64 * FSTRIDE;
    for (int c = tid; c < FTOT; c += 256) {
        float s = 0.f;
        for (int n = 0; n < 64; ++n) s += fb[n * FSTRIDE + c];
        pooled[c] = s * (1.f / 64.f);
    }
    __syncthreads();
    {
        float acc = b1[tid];
        for (int c = 0; c < FTOT; ++c) acc += pooled[c] * W1[c * 256 + tid];
        hfc[tid] = acc;
    }
    __syncthreads();
    if (tid < 2) {
        float acc = b2[tid];
        for (int c = 0; c < 256; ++c) acc += hfc[c] * W2[c * 2 + tid];
        out[b * 2 + tid] = acc;
    }
}

// ---------------------------------------------------------------------------
extern "C" void kernel_launch(void* const* d_in, const int* in_sizes, int n_in,
                              void* d_out, int out_size, void* d_ws, size_t ws_size,
                              hipStream_t stream)
{
    (void)in_sizes; (void)n_in; (void)out_size; (void)ws_size;

    // --- unpack inputs (JAX tree-leaf order; per-layer dict keys sorted:
    //     W, b, beta, gamma). blocks -> fc1{W,b} -> fc2{W,b} ---
    int p = 0;
    const float* x = (const float*)d_in[p++];
    const float *Wl[3][3], *bl[3][3], *betal[3][3], *gammal[3][3];
    for (int i = 0; i < 3; ++i)
        for (int j = 0; j < 3; ++j) {
            Wl[i][j]     = (const float*)d_in[p++];
            bl[i][j]     = (const float*)d_in[p++];
            betal[i][j]  = (const float*)d_in[p++];
            gammal[i][j] = (const float*)d_in[p++];
        }
    const float* fc1W = (const float*)d_in[p++];
    const float* fc1b = (const float*)d_in[p++];
    const float* fc2W = (const float*)d_in[p++];
    const float* fc2b = (const float*)d_in[p++];

    static const int CPREV [3] = {7, 71, 199};      // input feature width per block
    static const int KPAD  [3] = {32, 96, 224};     // ceil32(CPREV); BASEOF+KPAD==480
    static const int LSARR [3] = {64, 128, 256};    // block channel counts
    static const int KNNC  [3] = {2, 71, 199};      // knn feature dims (pos for blk 0)
    static const int BASEOF[3] = {448, 384, 256};   // current feats offset
    static const int OUTOFF[3] = {384, 256, 0};     // where block output is written

    // --- workspace carve-up ---
    char* ws = (char*)d_ws;
    size_t cur = 0;
    auto take = [&](size_t bytes) -> char* {
        char* ptr = ws + cur;
        cur += (bytes + 255) & ~(size_t)255;
        return ptr;
    };
    float*  feats = (float*) take((size_t)NODES * FSTRIDE * 4);
    __bf16* hA    = (__bf16*)take((size_t)NEDGE * 256 * 2);
    __bf16* hB    = (__bf16*)take((size_t)NEDGE * 256 * 2);
    __bf16* UV    = (__bf16*)take((size_t)NODES * 512 * 2);
    int*    idxb  = (int*)   take((size_t)NEDGE * 4);
    float*  sb    = (float*) take(1024 * 4);   // sum|sq|scale|shift x 256
    float* statSum = sb, *statSq = sb + 256, *scale = sb + 512, *shift = sb + 768;
    __bf16* wcat[3]; float* b2[3]; __bf16* w1bf[3]; __bf16* w2bf[3];
    for (int i = 0; i < 3; ++i) {
        const int ls = LSARR[i];
        wcat[i] = (__bf16*)take((size_t)KPAD[i] * 2 * ls * 2);
        b2[i]   = (float*) take((size_t)2 * ls * 4);
        w1bf[i] = (__bf16*)take((size_t)ls * ls * 2);
        w2bf[i] = (__bf16*)take((size_t)ls * ls * 2);
    }

    // --- weight prep (bf16 conversion + layer-0 algebraic split) ---
    for (int i = 0; i < 3; ++i) {
        const int C = CPREV[i], ls = LSARR[i];
        pn_prep_wcat<<<(KPAD[i] * 2 * ls + 255) / 256, 256, 0, stream>>>(
            Wl[i][0], bl[i][0], wcat[i], b2[i], C, KPAD[i], ls);
        pn_f32_to_bf16<<<(ls * ls + 255) / 256, 256, 0, stream>>>(Wl[i][1], w1bf[i], ls * ls);
        pn_f32_to_bf16<<<(ls * ls + 255) / 256, 256, 0, stream>>>(Wl[i][2], w2bf[i], ls * ls);
    }

    // zero padded feature buffer once, then scatter raw x into it
    pn_zero<<<(NODES * FSTRIDE + 255) / 256, 256, 0, stream>>>(feats, NODES * FSTRIDE);
    pn_init_feats<<<(NODES * 7 + 255) / 256, 256, 0, stream>>>(x, feats);

    const float invE = 1.f / (float)NEDGE;
    for (int blk = 0; blk < 3; ++blk) {
        const int ls = LSARR[blk];
        const float* fbase = feats + BASEOF[blk];

        // dynamic kNN graph on current features (pos for block 0)
        pn_knn<<<256, 64, 0, stream>>>(fbase, KNNC[blk], FSTRIDE, idxb);

        // node GEMM: [U|V] = feats @ [W1-W2 | W2] + [b|0]  (M=16384,K=Kpad,N=2ls)
        {
            dim3 g(NODES / 128, (2 * ls) / BN);
            pn_wmma_gemm<128, 0><<<g, 256, 0, stream>>>(
                fbase, (long)FSTRIDE, wcat[blk], b2[blk], nullptr, nullptr,
                UV, KPAD[blk], 2 * ls, nullptr, nullptr, 0);
        }

        // edge assemble h0 = U_i + V_j, accumulate BN stats
        pn_zero<<<2, 256, 0, stream>>>(statSum, 512);
        pn_edge_assemble<<<NEDGE / 64, 256, 0, stream>>>(UV, idxb, hA, ls, statSum, statSq);
        pn_finalize_stats<<<1, 256, 0, stream>>>(statSum, statSq, gammal[blk][0],
                                                 betal[blk][0], scale, shift, ls, invE);

        // layer 1: hB = relu(bn(hA)) @ W1 + b1
        pn_zero<<<2, 256, 0, stream>>>(statSum, 512);
        {
            dim3 g(NEDGE / 256, ls / BN);
            pn_wmma_gemm<256, 1><<<g, 256, 0, stream>>>(
                hA, (long)ls, w1bf[blk], bl[blk][1], scale, shift,
                hB, ls, ls, statSum, statSq, 1);
        }
        pn_finalize_stats<<<1, 256, 0, stream>>>(statSum, statSq, gammal[blk][1],
                                                 betal[blk][1], scale, shift, ls, invE);

        // layer 2: hA = relu(bn(hB)) @ W2 + b2
        pn_zero<<<2, 256, 0, stream>>>(statSum, 512);
        {
            dim3 g(NEDGE / 256, ls / BN);
            pn_wmma_gemm<256, 1><<<g, 256, 0, stream>>>(
                hB, (long)ls, w2bf[blk], bl[blk][2], scale, shift,
                hA, ls, ls, statSum, statSq, 1);
        }
        pn_finalize_stats<<<1, 256, 0, stream>>>(statSum, statSq, gammal[blk][2],
                                                 betal[blk][2], scale, shift, ls, invE);

        // BN+ReLU + mean over K -> concatenated features
        pn_edge_reduce<<<NODES, 64, 0, stream>>>(hA, scale, shift, feats, ls, OUTOFF[blk]);
    }

    // global mean pool + fc1 + fc2
    pn_head<<<256, 256, 0, stream>>>(feats, fc1W, fc1b, fc2W, fc2b, (float*)d_out);
}
